// SwitchMoE_87694642250249
// MI455X (gfx1250) — matrix operand order repaired and checked by reading, not compiled, not run on hardware
//
#include <hip/hip_runtime.h>

// ---------------- problem constants ----------------
#define T_TOK 8192      // B*S tokens
#define HID   1024
#define FFN_  4096
#define NE    8

// ---------------- GEMM tiling ----------------
#define BM 128
#define BN 128
#define BK 32
#define KP 20           // padded LDS row stride in uints (80B rows -> every 16B chunk aligned)

typedef __attribute__((ext_vector_type(16))) __bf16 v16bf;
typedef __attribute__((ext_vector_type(8)))  float  v8f;

union FragAB { v16bf v; unsigned int u[8]; };

#if __has_builtin(__builtin_amdgcn_global_load_async_to_lds_b128) && \
    __has_builtin(__builtin_amdgcn_s_wait_asynccnt)
#define USE_ASYNC 1
typedef int v4i32 __attribute__((vector_size(16)));
typedef __attribute__((address_space(1))) v4i32* gv4p;   // global int4*
typedef __attribute__((address_space(3))) v4i32* lv4p;   // LDS int4*
#else
#define USE_ASYNC 0
#endif

__device__ __forceinline__ unsigned short f32_to_bf16(float f) {
  unsigned int u = __float_as_uint(f);
  unsigned int r = (u + 0x7FFFu + ((u >> 16) & 1u)) >> 16;
  return (unsigned short)r;
}

// ---------------- elementwise fp32 -> bf16 (x) ----------------
__global__ __launch_bounds__(256) void convert_bf16_vec4(
    const float* __restrict__ src, unsigned short* __restrict__ dst, int n) {
  int i = (blockIdx.x * 256 + threadIdx.x) * 4;
  if (i >= n) return;
  float4 f = *(const float4*)(src + i);
  ushort4 o;
  o.x = f32_to_bf16(f.x); o.y = f32_to_bf16(f.y);
  o.z = f32_to_bf16(f.z); o.w = f32_to_bf16(f.w);
  *(ushort4*)(dst + i) = o;
}

// ---------------- transpose + convert: src [E][R][C] f32 -> dst [E][C][R] bf16 ----------------
__global__ __launch_bounds__(256) void transpose_convert_bf16(
    const float* __restrict__ src, unsigned short* __restrict__ dst, int R, int C) {
  __shared__ float tile[32][33];
  int e  = blockIdx.z;
  int r0 = blockIdx.y * 32, c0 = blockIdx.x * 32;
  const float* s = src + (size_t)e * R * C;
  unsigned short* d = dst + (size_t)e * R * C;
  int tx = threadIdx.x & 31, ty = threadIdx.x >> 5;   // 32 x 8
  #pragma unroll
  for (int i = 0; i < 32; i += 8)
    tile[ty + i][tx] = s[(size_t)(r0 + ty + i) * C + (c0 + tx)];
  __syncthreads();
  #pragma unroll
  for (int i = 0; i < 32; i += 8)
    d[(size_t)(c0 + ty + i) * R + (r0 + tx)] = f32_to_bf16(tile[tx][ty + i]);
}

// ---------------- gate: logits + top-2 (softmax is monotone; mask is binary) ----------------
__global__ __launch_bounds__(256) void gate_topk(
    const float* __restrict__ x, const float* __restrict__ gw, const float* __restrict__ gb,
    int* __restrict__ tok_e, int* __restrict__ counts) {
  int gtid = blockIdx.x * 256 + threadIdx.x;
  int t    = gtid >> 5;            // one wave32 per token
  int lane = threadIdx.x & 31;
  if (t >= T_TOK) return;
  float acc[NE];
  #pragma unroll
  for (int e = 0; e < NE; ++e) acc[e] = 0.f;
  const float* xr = x + (size_t)t * HID;
  for (int h = lane; h < HID; h += 32) {
    float xv = xr[h];
    const float* g = gw + h * NE;
    #pragma unroll
    for (int e = 0; e < NE; ++e) acc[e] += xv * g[e];
  }
  #pragma unroll
  for (int e = 0; e < NE; ++e) {
    #pragma unroll
    for (int off = 16; off > 0; off >>= 1) acc[e] += __shfl_xor(acc[e], off, 32);
  }
  if (lane == 0) {
    float m1 = -3.4e38f, m2 = -3.4e38f; int i1 = 0, i2 = 1;
    #pragma unroll
    for (int e = 0; e < NE; ++e) {
      float v = acc[e] + gb[e];
      if (v > m1)      { m2 = m1; i2 = i1; m1 = v; i1 = e; }
      else if (v > m2) { m2 = v; i2 = e; }
    }
    tok_e[t * 2 + 0] = i1;
    tok_e[t * 2 + 1] = i2;
    atomicAdd(&counts[i1], 1);
    atomicAdd(&counts[i2], 1);
  }
}

// ---------------- scan counts -> offsets, zero cursors ----------------
__global__ void scan_offsets(const int* __restrict__ counts, int* __restrict__ offsets,
                             int* __restrict__ cursor) {
  if (threadIdx.x == 0 && blockIdx.x == 0) {
    int s = 0;
    for (int e = 0; e < NE; ++e) { offsets[e] = s; s += counts[e]; cursor[e] = 0; }
    offsets[NE] = s;
  }
}

// ---------------- fill per-expert token lists ----------------
__global__ __launch_bounds__(256) void fill_assign(
    const int* __restrict__ tok_e, const int* __restrict__ offsets,
    int* __restrict__ cursor, int* __restrict__ assign_tok) {
  int t = blockIdx.x * 256 + threadIdx.x;
  if (t >= T_TOK) return;
  #pragma unroll
  for (int k = 0; k < 2; ++k) {
    int e = tok_e[t * 2 + k];
    int slot = atomicAdd(&cursor[e], 1);
    assign_tok[offsets[e] + slot] = t;
  }
}

// ============ unified expert GEMM ============
// PHASE1:  h1[a,:]   = relu(x[tok(a)] @ w1[e] + b1[e])   A=xb gathered, KDIM=HID,  NDIM=FFN
// PHASE2:  out[tok]  += h1[a] @ w2[e] + b2[e]            A=h1 direct,   KDIM=FFN,  NDIM=HID
template<int KDIM, int NDIM, bool PHASE1>
__global__ __launch_bounds__(256) void moe_gemm(
    const unsigned short* __restrict__ A,     // phase1: xb [T][KDIM]; phase2: h1 [2T+pad][KDIM]
    const unsigned short* __restrict__ BT,    // [E][NDIM][KDIM] (K-contiguous per output col)
    const float* __restrict__ bias,           // [E][NDIM]
    unsigned short* __restrict__ h1out,       // phase1 output (bf16, assignment-major)
    float* __restrict__ out,                  // phase2 output (f32, token-major, atomic)
    const int* __restrict__ assign_tok,       // [2T]
    const int* __restrict__ offsets) {        // [E+1]
  int e    = blockIdx.z;
  int base = offsets[e];
  int cnt  = offsets[e + 1] - base;
  int m0   = blockIdx.y * BM;
  if (m0 >= cnt) return;
  int n0   = blockIdx.x * BN;

  __shared__ unsigned int As[2][BM * KP];
  __shared__ unsigned int Bs[2][BN * KP];

  int tid  = threadIdx.x;
  int lane = tid & 31;
  int wid  = tid >> 5;
  int wm   = wid & 3;       // 4 waves along M -> 32 rows each
  int wn   = wid >> 2;      // 2 waves along N -> 64 cols each

  // ---- per-thread staging tasks (k-invariant: hoisted out of mainloop) ----
  const unsigned short* aSrc[2];
  const unsigned short* bSrc[2];
  int aOff[2], bOff[2];
  #pragma unroll
  for (int it = 0; it < 2; ++it) {
    int idx = tid + it * 256;
    int row = idx >> 2, seg = idx & 3;      // 128 rows x 4 chunks of 8 halfs
    int m   = m0 + row;
    int mm  = (m < cnt) ? m : (cnt - 1);
    size_t arow;
    if (PHASE1) arow = (size_t)assign_tok[base + mm] * KDIM;   // gather token row
    else        arow = (size_t)(base + mm) * KDIM;             // direct assignment row
    aSrc[it] = A + arow + seg * 8;
    aOff[it] = row * KP + seg * 4;
    bSrc[it] = BT + ((size_t)e * NDIM + (n0 + row)) * KDIM + seg * 8;
    bOff[it] = row * KP + seg * 4;
  }

  v8f acc[2][4];
  #pragma unroll
  for (int i = 0; i < 2; ++i)
    #pragma unroll
    for (int j = 0; j < 4; ++j) acc[i][j] = (v8f){0.f,0.f,0.f,0.f,0.f,0.f,0.f,0.f};

  int l16 = lane & 15;
  int akb = (lane >> 4) * 4;   // A frag: K0-7/K16-23 vs K8-15/K24-31 uint groups
  int bkb = (lane >> 4) * 4;   // B frag: K0-7/K16-23 vs K8-15/K24-31 uint groups

  auto compute = [&](const unsigned int* __restrict__ Ab, const unsigned int* __restrict__ Bb) {
    FragAB a[2], b[4];
    #pragma unroll
    for (int i = 0; i < 2; ++i) {
      const unsigned int* s = &Ab[(wm * 32 + i * 16 + l16) * KP];
      #pragma unroll
      for (int q = 0; q < 4; ++q) { a[i].u[q] = s[akb + q]; a[i].u[4 + q] = s[akb + 8 + q]; }
    }
    #pragma unroll
    for (int j = 0; j < 4; ++j) {
      const unsigned int* sb = &Bb[(wn * 64 + j * 16 + l16) * KP];
      #pragma unroll
      for (int q = 0; q < 8; ++q) b[j].u[q] = sb[bkb + (q & 3) + (q >> 2) * 8];
    }
    #pragma unroll
    for (int i = 0; i < 2; ++i)
      #pragma unroll
      for (int j = 0; j < 4; ++j)
        acc[i][j] = __builtin_amdgcn_wmma_f32_16x16x32_bf16(
            false, a[i].v, false, b[j].v, (short)0, acc[i][j], false, false);
  };

#if USE_ASYNC
  // -------- async-to-LDS double-buffered mainloop (ASYNCcnt) --------
  auto stage = [&](int k0, int buf) {
    #pragma unroll
    for (int it = 0; it < 2; ++it) {
      __builtin_amdgcn_global_load_async_to_lds_b128(
          (gv4p)(aSrc[it] + k0), (lv4p)&As[buf][aOff[it]], 0, 0);
      __builtin_amdgcn_global_load_async_to_lds_b128(
          (gv4p)(bSrc[it] + k0), (lv4p)&Bs[buf][bOff[it]], 0, 0);
    }
  };
  stage(0, 0);
  __builtin_amdgcn_s_wait_asynccnt(0);
  __syncthreads();
  int cur = 0;
  for (int k0 = BK; k0 < KDIM; k0 += BK) {
    stage(k0, cur ^ 1);                 // async fill of next buffer
    compute(As[cur], Bs[cur]);          // overlap WMMA with async copies
    __builtin_amdgcn_s_wait_asynccnt(0);
    __syncthreads();
    cur ^= 1;
  }
  compute(As[cur], Bs[cur]);
#else
  // -------- register-staged double-buffered mainloop --------
  uint4 ra[2], rb[2];
  auto loadRegs = [&](int k0) {
    #pragma unroll
    for (int it = 0; it < 2; ++it) {
      ra[it] = *(const uint4*)(aSrc[it] + k0);
      rb[it] = *(const uint4*)(bSrc[it] + k0);
      if (k0 + BK < KDIM) {
        __builtin_prefetch(aSrc[it] + k0 + BK, 0, 3);
        __builtin_prefetch(bSrc[it] + k0 + BK, 0, 3);
      }
    }
  };
  auto storeRegs = [&](int buf) {
    #pragma unroll
    for (int it = 0; it < 2; ++it) {
      unsigned int* da = &As[buf][aOff[it]];
      da[0] = ra[it].x; da[1] = ra[it].y; da[2] = ra[it].z; da[3] = ra[it].w;
      unsigned int* db = &Bs[buf][bOff[it]];
      db[0] = rb[it].x; db[1] = rb[it].y; db[2] = rb[it].z; db[3] = rb[it].w;
    }
  };
  loadRegs(0);
  storeRegs(0);
  __syncthreads();
  int cur = 0;
  for (int k0 = BK; k0 < KDIM; k0 += BK) {
    loadRegs(k0);                       // global loads in flight
    compute(As[cur], Bs[cur]);          // overlap WMMA with loads
    storeRegs(cur ^ 1);
    __syncthreads();
    cur ^= 1;
  }
  compute(As[cur], Bs[cur]);
#endif

  // -------- epilogue --------
  bool full = (m0 + BM) <= cnt;         // uniform fast path: no per-element bounds checks
  int half  = lane >> 4;
  #pragma unroll
  for (int i = 0; i < 2; ++i)
    #pragma unroll
    for (int j = 0; j < 4; ++j) {
      int rbase = m0 + wm * 32 + i * 16 + half * 8;
      int c     = n0 + wn * 64 + j * 16 + l16;
      float bsv = bias[e * NDIM + c];
      #pragma unroll
      for (int v = 0; v < 8; ++v) {
        int m = rbase + v;
        if (full || m < cnt) {
          float val = acc[i][j][v] + bsv;
          if constexpr (PHASE1) {
            val = val > 0.f ? val : 0.f;
            h1out[(size_t)(base + m) * NDIM + c] = f32_to_bf16(val);
          } else {
            int tok = assign_tok[base + m];
            atomicAdd(&out[(size_t)tok * NDIM + c], val);
          }
        }
      }
    }
}

// ---------------- host orchestration ----------------
extern "C" void kernel_launch(void* const* d_in, const int* in_sizes, int n_in,
                              void* d_out, int out_size, void* d_ws, size_t ws_size,
                              hipStream_t stream) {
  const float* x  = (const float*)d_in[0];   // [T][HID]
  const float* gw = (const float*)d_in[1];   // [HID][NE]
  const float* gb = (const float*)d_in[2];   // [NE]
  const float* w1 = (const float*)d_in[3];   // [NE][HID][FFN]
  const float* b1 = (const float*)d_in[4];   // [NE][FFN]
  const float* w2 = (const float*)d_in[5];   // [NE][FFN][HID]
  const float* b2 = (const float*)d_in[6];   // [NE][HID]
  float* out = (float*)d_out;                // [T][HID]

  char* ws = (char*)d_ws;
  size_t off = 0;
  auto take = [&](size_t bytes) -> char* {
    char* p = ws + off;
    off = (off + bytes + 255) & ~(size_t)255;
    return p;
  };
  unsigned short* xb   = (unsigned short*)take((size_t)T_TOK * HID * 2);
  unsigned short* w1T  = (unsigned short*)take((size_t)NE * HID * FFN_ * 2);
  unsigned short* w2T  = (unsigned short*)take((size_t)NE * HID * FFN_ * 2);
  unsigned short* h1   = (unsigned short*)take(((size_t)2 * T_TOK + BM) * FFN_ * 2);
  int* tok_e      = (int*)take((size_t)T_TOK * 2 * 4);
  int* assign_tok = (int*)take((size_t)2 * T_TOK * 4);
  int* counts     = (int*)take(32 * 4);   // counts[8] | offsets[9] | cursor[8]
  int* offsets    = counts + 8;
  int* cursor     = counts + 20;

  // zero routing counters and output accumulator
  (void)hipMemsetAsync(counts, 0, 32 * 4, stream);
  (void)hipMemsetAsync(out, 0, (size_t)T_TOK * HID * 4, stream);

  // fp32 -> bf16 conversions
  {
    int n = T_TOK * HID;
    convert_bf16_vec4<<<(n / 4 + 255) / 256, 256, 0, stream>>>(x, xb, n);
  }
  transpose_convert_bf16<<<dim3(FFN_ / 32, HID / 32, NE), 256, 0, stream>>>(w1, w1T, HID, FFN_);
  transpose_convert_bf16<<<dim3(HID / 32, FFN_ / 32, NE), 256, 0, stream>>>(w2, w2T, FFN_, HID);

  // gating + routing
  gate_topk<<<(T_TOK * 32) / 256, 256, 0, stream>>>(x, gw, gb, tok_e, counts);
  scan_offsets<<<1, 32, 0, stream>>>(counts, offsets, cursor);
  fill_assign<<<T_TOK / 256, 256, 0, stream>>>(tok_e, offsets, cursor, assign_tok);

  // expert GEMMs (worst-case grid in M; blocks early-exit past each expert's count)
  moe_gemm<HID, FFN_, true><<<dim3(FFN_ / BN, T_TOK / BM, NE), 256, 0, stream>>>(
      xb, w1T, b1, h1, nullptr, assign_tok, offsets);
  moe_gemm<FFN_, HID, false><<<dim3(HID / BN, T_TOK / BM, NE), 256, 0, stream>>>(
      h1, w2T, b2, nullptr, out, assign_tok, offsets);
}
